// CharRNN_17111149707452
// MI455X (gfx1250) — matrix operand order repaired and checked by reading, not compiled, not run on hardware
//
#include <hip/hip_runtime.h>
#include <math.h>

// Problem sizes (fixed by the reference)
#define T_STEPS 16384
#define E_DIM   1024
#define H_DIM   1024
#define G_DIM   4096      // 4*H (gate rows, torch order i,f,g,o)

// Persistent scan configuration
#define NWG         128   // persistent workgroups (each ~136KB LDS -> 2 per WGP)
#define HID_PER_WG  8     // hidden units owned per WG
#define ROWS_PER_WG 32    // 4 gates * HID_PER_WG rows of W_hh
#define W_STRIDE    1028  // padded LDS row stride in floats (bank de-phasing)

typedef _Float16 v4h  __attribute__((ext_vector_type(4)));
typedef _Float16 v16h __attribute__((ext_vector_type(16)));
typedef float    v8f  __attribute__((ext_vector_type(8)));

// ---------------------------------------------------------------------------
// Fragment-order LDS offsets for V_WMMA_F32_16X16X32_F16 (ISA 7.12.2):
//  A 16x32: element (m,k) -> frag lane m + 16*((k>>3)&1), half (k&7) + 8*(k>>4)
//  B 32x16: element (k,n) -> frag lane n + 16*((k>>3)&1), half (k&7) + 8*(k>>4)
// Index = frag_lane*16 + half, so each lane reads its v16h contiguously.
// ---------------------------------------------------------------------------
__device__ __forceinline__ int a_off(int m, int k) {
  return ((m + (((k >> 3) & 1) << 4)) << 4) + (k & 7) + ((k >> 4) << 3);
}
__device__ __forceinline__ int b_off(int n, int k) {
  return ((n + (((k >> 3) & 1) << 4)) << 4) + (k & 7) + ((k >> 4) << 3);
}

__device__ __forceinline__ void split4(const float4 x, v4h* hi, v4h* lo) {
  _Float16 h0 = (_Float16)x.x, h1 = (_Float16)x.y,
           h2 = (_Float16)x.z, h3 = (_Float16)x.w;
  *hi = (v4h){h0, h1, h2, h3};
  *lo = (v4h){(_Float16)(x.x - (float)h0), (_Float16)(x.y - (float)h1),
              (_Float16)(x.z - (float)h2), (_Float16)(x.w - (float)h3)};
}

// ---------------------------------------------------------------------------
// Kernel 1: xg[T,4H] = Xs[T,E] @ Wih[4H,E]^T + (bih + bhh)
// One wave per 16x64 output strip (4 accumulators; A-frag reused 4x).
// K in chunks of 32; split-f16 hi/lo so the f32 result is ~fp32-accurate:
// D += Ah*Bh + Ah*Bl + Al*Bh  (12 WMMAs per chunk).
// ---------------------------------------------------------------------------
__global__ __launch_bounds__(32) void xproj_gemm(
    const float* __restrict__ Xs, const float* __restrict__ Wih,
    const float* __restrict__ bih, const float* __restrict__ bhh,
    float* __restrict__ xg) {
  __shared__ __align__(32) _Float16 AsHi[512];
  __shared__ __align__(32) _Float16 AsLo[512];
  __shared__ __align__(32) _Float16 BsHi[4 * 512];   // 4 N-subtiles
  __shared__ __align__(32) _Float16 BsLo[4 * 512];

  const int lane = threadIdx.x;           // 0..31
  const int m0 = blockIdx.y << 4;         // timestep tile (16 rows)
  const int n0 = blockIdx.x << 6;         // gate-row strip (64 cols)

  v8f acc0 = {0.f, 0.f, 0.f, 0.f, 0.f, 0.f, 0.f, 0.f};
  v8f acc1 = acc0, acc2 = acc0, acc3 = acc0;

  for (int kc = 0; kc < E_DIM; kc += 32) {
    // --- Stage A tile (16x32): 128 float4 -> 4 per lane ---
#pragma unroll
    for (int j = 0; j < 4; ++j) {
      const int f    = lane + (j << 5);   // 0..127
      const int mrow = f >> 3;            // 0..15
      const int k0   = (f & 7) << 2;      // 0,4,...,28
      const float4 xa =
          *(const float4*)(Xs + (size_t)(m0 + mrow) * E_DIM + kc + k0);
      v4h hi, lo;
      split4(xa, &hi, &lo);
      *(v4h*)(AsHi + a_off(mrow, k0)) = hi;
      *(v4h*)(AsLo + a_off(mrow, k0)) = lo;
    }
    // --- Stage B strip (64x32): 512 float4 -> 16 per lane ---
#pragma unroll
    for (int j = 0; j < 16; ++j) {
      const int f    = lane + (j << 5);   // 0..511
      const int nrow = f >> 3;            // 0..63
      const int k0   = (f & 7) << 2;      // 0,4,...,28
      const int s    = nrow >> 4;         // subtile 0..3
      const int n    = nrow & 15;
      const float4 xb =
          *(const float4*)(Wih + (size_t)(n0 + nrow) * E_DIM + kc + k0);
      v4h hi, lo;
      split4(xb, &hi, &lo);
      *(v4h*)(BsHi + (s << 9) + b_off(n, k0)) = hi;
      *(v4h*)(BsLo + (s << 9) + b_off(n, k0)) = lo;
    }
    __syncthreads();

    const v16h ah = *(const v16h*)(AsHi + (lane << 4));
    const v16h al = *(const v16h*)(AsLo + (lane << 4));
#pragma unroll
    for (int s = 0; s < 4; ++s) {
      const v16h bh = *(const v16h*)(BsHi + (s << 9) + (lane << 4));
      const v16h bl = *(const v16h*)(BsLo + (s << 9) + (lane << 4));
      v8f a = (s == 0) ? acc0 : (s == 1) ? acc1 : (s == 2) ? acc2 : acc3;
      a = __builtin_amdgcn_wmma_f32_16x16x32_f16(false, ah, false, bh,
                                                 (short)0, a, false, false);
      a = __builtin_amdgcn_wmma_f32_16x16x32_f16(false, ah, false, bl,
                                                 (short)0, a, false, false);
      a = __builtin_amdgcn_wmma_f32_16x16x32_f16(false, al, false, bh,
                                                 (short)0, a, false, false);
      if (s == 0) acc0 = a; else if (s == 1) acc1 = a;
      else if (s == 2) acc2 = a; else acc3 = a;
    }
    __syncthreads();
  }

  // Epilogue: D layout -> VGPR v: lanes 0-15 M=v, lanes 16-31 M=v+8; N=lane&15
  const int col   = lane & 15;
  const int mbase = m0 + ((lane >> 4) << 3);
#pragma unroll
  for (int s = 0; s < 4; ++s) {
    const v8f a = (s == 0) ? acc0 : (s == 1) ? acc1 : (s == 2) ? acc2 : acc3;
    const int nc = n0 + (s << 4) + col;
    const float bias = bih[nc] + bhh[nc];
#pragma unroll
    for (int v = 0; v < 8; ++v) {
      xg[(size_t)(mbase + v) * G_DIM + nc] = a[v] + bias;
    }
  }
}

// ---------------------------------------------------------------------------
// Grid-wide sense barrier (persistent kernel). cnt/gen zeroed per launch via
// hipMemsetAsync, so targets are simply step+1.
// ---------------------------------------------------------------------------
__device__ __forceinline__ void grid_sync(unsigned* cnt, unsigned* gen,
                                          unsigned target) {
  __threadfence();
  __syncthreads();
  if (threadIdx.x == 0) {
    if (atomicAdd(cnt, 1u) == NWG - 1) {
      atomicExch(cnt, 0u);
      __threadfence();
      atomicAdd(gen, 1u);
    } else {
      while (__hip_atomic_load(gen, __ATOMIC_ACQUIRE,
                               __HIP_MEMORY_SCOPE_AGENT) < target) {
        __builtin_amdgcn_s_sleep(2);
      }
    }
  }
  __syncthreads();
  __threadfence();
}

// ---------------------------------------------------------------------------
// Kernel 2: persistent LSTM scan. Each WG pins its 32 rows of W_hh in LDS
// (rows {q*H + wg*8 + i}), computes its gate slice each step, updates its 8
// (h,c) pairs locally, publishes h through global Hout, grid-syncs.
// ---------------------------------------------------------------------------
__global__ __launch_bounds__(256) void lstm_scan(
    const float* __restrict__ Whh, const float* __restrict__ xg,
    float* __restrict__ Hout, unsigned* __restrict__ bar) {
  __shared__ float Wl[ROWS_PER_WG * W_STRIDE];  // 131,584 B
  __shared__ float hl[H_DIM];                   // 4 KB: h_{t-1}
  __shared__ float gl[ROWS_PER_WG];             // gate pre-activations
  __shared__ float cl[HID_PER_WG];              // cell state

  const int tid = threadIdx.x;
  const int wg  = blockIdx.x;

  // --- Load this WG's W_hh slice into LDS (coalesced float4) ---
  for (int n = tid; n < ROWS_PER_WG * 256; n += 256) {
    const int lr  = n >> 8;                 // local row 0..31
    const int col = (n & 255) << 2;         // 0..1020
    const int q   = lr >> 3;                // gate 0..3
    const int i   = lr & 7;                 // hidden unit 0..7
    const int grow = (q << 10) + (wg << 3) + i;
    *(float4*)(Wl + lr * W_STRIDE + col) =
        *(const float4*)(Whh + (size_t)grow * H_DIM + col);
  }
  if (tid < HID_PER_WG) cl[tid] = 0.f;
  __syncthreads();

  const int r = tid >> 3;                   // local row this thread helps with
  const int p = tid & 7;                    // column-chunk 0..7 (128 cols each)
  const int q = r >> 3;
  const int i = r & 7;
  const int grow = (q << 10) + (wg << 3) + i;

  unsigned target = 0;
  for (int t = 0; t < T_STEPS; ++t) {
    // h_{t-1} into LDS (t==0: h0 = 0)
    if (t == 0) {
      ((float4*)hl)[tid] = make_float4(0.f, 0.f, 0.f, 0.f);
    } else {
      ((float4*)hl)[tid] =
          *(const float4*)(Hout + (size_t)(t - 1) * H_DIM + (tid << 2));
    }
    // Issue the xg gather early: independent of h, latency hides under dot.
    float xval = 0.f;
    if (p == 0) xval = xg[(size_t)t * G_DIM + grow];
    if (p == 0 && t + 1 < T_STEPS)
      __builtin_prefetch(xg + (size_t)(t + 1) * G_DIM + grow, 0, 1);
    __syncthreads();

    // Partial dot: 128 cols per thread, float4 from LDS
    const float* wrow = Wl + r * W_STRIDE + (p << 7);
    const float* hp   = hl + (p << 7);
    float s0 = 0.f, s1 = 0.f, s2 = 0.f, s3 = 0.f;
#pragma unroll 8
    for (int m = 0; m < 128; m += 4) {
      const float4 w  = *(const float4*)(wrow + m);
      const float4 h4 = *(const float4*)(hp + m);
      s0 = fmaf(w.x, h4.x, s0);
      s1 = fmaf(w.y, h4.y, s1);
      s2 = fmaf(w.z, h4.z, s2);
      s3 = fmaf(w.w, h4.w, s3);
    }
    float part = (s0 + s1) + (s2 + s3);
    part += __shfl_xor(part, 1);
    part += __shfl_xor(part, 2);
    part += __shfl_xor(part, 4);
    if (p == 0) gl[r] = part + xval;
    __syncthreads();

    // Gate nonlinearities + state update for the 8 owned hidden units
    if (tid < HID_PER_WG) {
      const float gi = gl[tid];                      // input gate row
      const float gf = gl[HID_PER_WG + tid];         // forget
      const float gg = gl[2 * HID_PER_WG + tid];     // candidate
      const float go = gl[3 * HID_PER_WG + tid];     // output
      const float ig = 1.f / (1.f + expf(-gi));
      const float fg = 1.f / (1.f + expf(-gf));
      const float gt = tanhf(gg);
      const float og = 1.f / (1.f + expf(-go));
      const float c  = fmaf(fg, cl[tid], ig * gt);
      cl[tid] = c;
      Hout[(size_t)t * H_DIM + (wg << 3) + tid] = og * tanhf(c);
    }

    ++target;
    grid_sync(bar, bar + 1, target);
  }
}

// ---------------------------------------------------------------------------
// Kernel 3: per-timestep cross-entropy on raw hidden output.
// loss[t] = max + log(sum exp(h-max)) - h[y_t]; deterministic tree reductions.
// ---------------------------------------------------------------------------
__global__ __launch_bounds__(256) void loss_rows(
    const float* __restrict__ Hout, const int* __restrict__ ys,
    float* __restrict__ losses) {
  __shared__ float red[256];
  const int t = blockIdx.x, tid = threadIdx.x;
  const float* h = Hout + (size_t)t * H_DIM;
  const float4 x = ((const float4*)h)[tid];

  float mx = fmaxf(fmaxf(x.x, x.y), fmaxf(x.z, x.w));
  red[tid] = mx;
  __syncthreads();
  for (int s = 128; s > 0; s >>= 1) {
    if (tid < s) red[tid] = fmaxf(red[tid], red[tid + s]);
    __syncthreads();
  }
  const float M = red[0];
  __syncthreads();

  float sum = expf(x.x - M) + expf(x.y - M) + expf(x.z - M) + expf(x.w - M);
  red[tid] = sum;
  __syncthreads();
  for (int s = 128; s > 0; s >>= 1) {
    if (tid < s) red[tid] += red[tid + s];
    __syncthreads();
  }
  if (tid == 0) losses[t] = M + logf(red[0]) - h[ys[t]];
}

// ---------------------------------------------------------------------------
// Kernel 4: deterministic final sum -> d_out[0]
// ---------------------------------------------------------------------------
__global__ __launch_bounds__(256) void loss_sum(
    const float* __restrict__ losses, float* __restrict__ out) {
  __shared__ float red[256];
  const int tid = threadIdx.x;
  float s = 0.f;
  for (int t = tid; t < T_STEPS; t += 256) s += losses[t];
  red[tid] = s;
  __syncthreads();
  for (int k = 128; k > 0; k >>= 1) {
    if (tid < k) red[tid] += red[tid + k];
    __syncthreads();
  }
  if (tid == 0) out[0] = red[0];
}

// ---------------------------------------------------------------------------
extern "C" void kernel_launch(void* const* d_in, const int* in_sizes, int n_in,
                              void* d_out, int out_size, void* d_ws, size_t ws_size,
                              hipStream_t stream) {
  const float* Xs  = (const float*)d_in[0];
  const float* Wih = (const float*)d_in[1];
  const float* Whh = (const float*)d_in[2];
  const float* bih = (const float*)d_in[3];
  const float* bhh = (const float*)d_in[4];
  const int*   ys  = (const int*)d_in[5];
  float* out = (float*)d_out;

  // Workspace layout (all 16B aligned): barrier | losses[T] | Hout[T*H] | xg[T*4H]
  char* ws = (char*)d_ws;
  unsigned* bar  = (unsigned*)ws;                                           // 256 B
  float* losses  = (float*)(ws + 256);                                      // 64 KB
  float* Hout    = (float*)(ws + 256 + (size_t)T_STEPS * 4);                // 64 MB
  float* xg      = (float*)(ws + 256 + (size_t)T_STEPS * 4 +
                            (size_t)T_STEPS * H_DIM * 4);                   // 256 MB

  (void)hipMemsetAsync(bar, 0, 256, stream);

  xproj_gemm<<<dim3(G_DIM / 64, T_STEPS / 16), 32, 0, stream>>>(
      Xs, Wih, bih, bhh, xg);
  lstm_scan<<<NWG, 256, 0, stream>>>(Whh, xg, Hout, bar);
  loss_rows<<<T_STEPS, 256, 0, stream>>>(Hout, ys, losses);
  loss_sum<<<1, 256, 0, stream>>>(losses, out);
}
